// GAT_GCN_g_mean_47107201303303
// MI455X (gfx1250) — compile-verified, matrix operand order checked by reading, not run on hardware
//
#include <hip/hip_runtime.h>
#include <hip/hip_bf16.h>

#define NODES_F0   78
#define HEADS      10
#define FH         780        // 78*10
#define BATCHES    128
#define LCONV      1017
#define CIN        640
#define KFLAT      5120       // 640*8
#define CONVFLAT   32544      // 32*1017
#define STAGE_K    64         // K per LDS stage (two 32-wide WMMA chunks)

typedef __attribute__((ext_vector_type(16))) __bf16 v16bf;
typedef __attribute__((ext_vector_type(8)))  __bf16 v8bf;
typedef __attribute__((ext_vector_type(8)))  float  v8f;

__device__ __forceinline__ v8f wmma_bf16(const v16bf& a, const v16bf& b, const v8f& c) {
    return __builtin_amdgcn_wmma_f32_16x16x32_bf16(false, a, false, b, (short)0, c, false, false);
}

// ---------------------------------------------------------------------------
// Operand packing: fp32 -> zero-padded bf16
// ---------------------------------------------------------------------------
__global__ void k_pack_a(const float* __restrict__ A, __bf16* __restrict__ Ap,
                         int M, int K, int Kp)
{
    size_t i = (size_t)blockIdx.x * blockDim.x + threadIdx.x;
    if (i >= (size_t)M * Kp) return;
    int r = (int)(i / Kp), c = (int)(i % Kp);
    Ap[i] = (__bf16)((c < K) ? A[(size_t)r * K + c] : 0.f);
}

__global__ void k_pack_bt(const float* __restrict__ B, __bf16* __restrict__ Btp,
                          int K, int N, int Np, int Kp)
{
    size_t i = (size_t)blockIdx.x * blockDim.x + threadIdx.x;
    if (i >= (size_t)Np * Kp) return;
    int n = (int)(i / Kp), k = (int)(i % Kp);
    Btp[i] = (__bf16)((n < N && k < K) ? B[(size_t)k * N + n] : 0.f);
}

// ---------------------------------------------------------------------------
// WMMA GEMM: C[M,N] = A @ B. A bf16 [M,Kp]; B pre-transposed bf16 Bt [Np,Kp],
// Kp % 64 == 0. 4 waves/block; block computes 64 rows x 32 cols.
// The shared 32 x 64 bf16 B stage (4 KB) is double-buffered in LDS and filled
// with the CDNA5 async global->LDS path (2x global_load_async_to_lds_b128 per
// thread per stage). The copy for stage i+1 is issued before consuming stage
// i and waited with s_wait_asynccnt 0x2 (in-order completion), overlapping
// DMA with 4 back-to-back WMMAs per stage. A streams via pipelined VMEM b128.
// epi: 0 bias, 1 bias->BN->ReLU, 2 bias->ReLU->BN.
// ---------------------------------------------------------------------------
__global__ void k_wmma_gemm(const __bf16* __restrict__ A,
                            const __bf16* __restrict__ Bt,
                            float* __restrict__ C,
                            int M, int Nn, int Kp, int Mtiles,
                            const float* __restrict__ bias,
                            const float* __restrict__ bnp, int epi)
{
    __shared__ __bf16 ldsB[2][32 * STAGE_K];     // ping-pong, 2 x 4 KB

    const int t    = threadIdx.x;                // 0..127
    const int lane = t & 31;
    const int wv   = t >> 5;
    int tm = blockIdx.y * 4 + wv;
    if (tm >= Mtiles) tm = Mtiles - 1;           // clamp (never skips barriers)
    const int tn   = blockIdx.x;                 // 32 columns per block
    const int g    = lane >> 4;
    const int lr   = lane & 15;

    const __bf16* ap   = A + (size_t)(tm * 16 + lr) * Kp;
    // thread's slice of the B stage: row t>>2, 16B segment t&3 (x2 chunks)
    const __bf16* bsrc = Bt + (size_t)(tn * 32 + (t >> 2)) * Kp + (t & 3) * 8;
    const unsigned rowOff = (unsigned)((t >> 2) * (STAGE_K * 2) + (t & 3) * 16);
    const unsigned lds0 = (unsigned)(size_t)(&ldsB[0][0]) + rowOff;
    const unsigned lds1 = (unsigned)(size_t)(&ldsB[1][0]) + rowOff;

#define ISSUE_STAGE(LBASE, KB)                                                 \
    do {                                                                       \
        asm volatile("global_load_async_to_lds_b128 %0, %1, off"               \
                     :: "v"(LBASE),                                            \
                        "v"((unsigned long long)(size_t)(bsrc + (KB)))         \
                     : "memory");                                              \
        asm volatile("global_load_async_to_lds_b128 %0, %1, off"               \
                     :: "v"(LBASE + 64u),                                      \
                        "v"((unsigned long long)(size_t)(bsrc + (KB) + 32))    \
                     : "memory");                                              \
    } while (0)

    v8f acc0 = {0.f,0.f,0.f,0.f,0.f,0.f,0.f,0.f};
    v8f acc1 = {0.f,0.f,0.f,0.f,0.f,0.f,0.f,0.f};

    ISSUE_STAGE(lds0, 0);
    int buf = 0;
    for (int kb = 0; kb < Kp; kb += STAGE_K) {
        const bool more = (kb + STAGE_K) < Kp;       // uniform
        if (more) ISSUE_STAGE(buf ? lds0 : lds1, kb + STAGE_K);

        // A fragments for both 32-K chunks (4x global b128)
        v8bf a0l = *(const v8bf*)(ap + kb + g * 8);
        v8bf a0h = *(const v8bf*)(ap + kb + g * 8 + 16);
        v8bf a1l = *(const v8bf*)(ap + kb + 32 + g * 8);
        v8bf a1h = *(const v8bf*)(ap + kb + 32 + g * 8 + 16);

        if (more) asm volatile("s_wait_asynccnt 0x2" ::: "memory");
        else      asm volatile("s_wait_asynccnt 0x0" ::: "memory");
        __syncthreads();

        const __bf16* lb = &ldsB[buf][0];            // row stride 64 bf16
        v8bf b00l = *(const v8bf*)(lb + lr * STAGE_K + g * 8);
        v8bf b00h = *(const v8bf*)(lb + lr * STAGE_K + g * 8 + 16);
        v8bf b10l = *(const v8bf*)(lb + (16 + lr) * STAGE_K + g * 8);
        v8bf b10h = *(const v8bf*)(lb + (16 + lr) * STAGE_K + g * 8 + 16);
        v8bf b01l = *(const v8bf*)(lb + lr * STAGE_K + 32 + g * 8);
        v8bf b01h = *(const v8bf*)(lb + lr * STAGE_K + 32 + g * 8 + 16);
        v8bf b11l = *(const v8bf*)(lb + (16 + lr) * STAGE_K + 32 + g * 8);
        v8bf b11h = *(const v8bf*)(lb + (16 + lr) * STAGE_K + 32 + g * 8 + 16);

        v16bf a0, a1, b00, b10, b01, b11;
#pragma unroll
        for (int i = 0; i < 8; ++i) {
            a0[i]  = a0l[i];  a0[8 + i]  = a0h[i];
            a1[i]  = a1l[i];  a1[8 + i]  = a1h[i];
            b00[i] = b00l[i]; b00[8 + i] = b00h[i];
            b10[i] = b10l[i]; b10[8 + i] = b10h[i];
            b01[i] = b01l[i]; b01[8 + i] = b01h[i];
            b11[i] = b11l[i]; b11[8 + i] = b11h[i];
        }
        acc0 = wmma_bf16(a0, b00, acc0);
        acc1 = wmma_bf16(a0, b10, acc1);
        acc0 = wmma_bf16(a1, b01, acc0);
        acc1 = wmma_bf16(a1, b11, acc1);

        __syncthreads();                             // readers done before reuse
        buf ^= 1;
    }
#undef ISSUE_STAGE

#pragma unroll
    for (int r = 0; r < 8; ++r) {
        const int row = tm * 16 + r + g * 8;
        if (row >= M) continue;
#pragma unroll
        for (int half = 0; half < 2; ++half) {
            const int col = tn * 32 + half * 16 + lr;
            if (col >= Nn) continue;
            float v = half ? acc1[r] : acc0[r];
            if (bias) v += bias[col];
            if (epi == 1) {
                float tv = bnp[col] * (v - bnp[2 * Nn + col]) *
                           rsqrtf(bnp[3 * Nn + col] + 1e-5f) + bnp[Nn + col];
                v = fmaxf(tv, 0.f);
            } else if (epi == 2) {
                v = fmaxf(v, 0.f);
                v = bnp[col] * (v - bnp[2 * Nn + col]) *
                    rsqrtf(bnp[3 * Nn + col] + 1e-5f) + bnp[Nn + col];
            }
            C[(size_t)row * Nn + col] = v;
        }
    }
}

// ---------------------------------------------------------------------------
// Conv1d(640->32, k=8) implicit WMMA GEMM. M=130176 (exact), N=32, K=5120.
// A run of 8 K-values = one contiguous filter-tap row target[b,c,l..l+7].
// Filter is pre-packed bf16 [32, 5120] (L2-resident). Writes bf16
// [B, ostride] (ostride = 64-padded 32*1017) for the fcxt GEMM.
// ---------------------------------------------------------------------------
__global__ void k_conv_wmma(const float* __restrict__ T,
                            const __bf16* __restrict__ Wt,
                            const float* __restrict__ bias,
                            __bf16* __restrict__ Out, int ostride)
{
    const int lane = threadIdx.x & 31;
    const int wv   = threadIdx.x >> 5;
    const int tm   = blockIdx.x * 4 + wv;    // 0..8135
    const int g    = lane >> 4;
    const int lr   = lane & 15;
    const int m    = tm * 16 + lr;           // b*1017 + l
    const int bb   = m / LCONV;
    const int l    = m % LCONV;

    const __bf16* wp0 = Wt + (size_t)lr * KFLAT;
    const __bf16* wp1 = Wt + (size_t)(16 + lr) * KFLAT;
    const float*  tb  = T + (size_t)bb * (CIN * 1024) + l;

    v8f acc0 = {0.f,0.f,0.f,0.f,0.f,0.f,0.f,0.f};
    v8f acc1 = {0.f,0.f,0.f,0.f,0.f,0.f,0.f,0.f};

    for (int kb = 0; kb < KFLAT; kb += 32) {
        const int o  = kb + g * 8;           // multiple of 8 -> one c row
        const int c0 = o >> 3;
        const float* a0 = tb + (size_t)c0 * 1024;        // 8 contiguous f32
        const float* a1 = a0 + 2 * 1024;                 // c0+2
        if (kb + 32 < KFLAT) __builtin_prefetch(a0 + 4 * 1024, 0, 0);
        v8bf w0l = *(const v8bf*)(wp0 + o);
        v8bf w0h = *(const v8bf*)(wp0 + o + 16);
        v8bf w1l = *(const v8bf*)(wp1 + o);
        v8bf w1h = *(const v8bf*)(wp1 + o + 16);
        v16bf a, b0, b1;
#pragma unroll
        for (int i = 0; i < 8; ++i) {
            a[i]      = (__bf16)a0[i];
            a[8 + i]  = (__bf16)a1[i];
            b0[i] = w0l[i]; b0[8 + i] = w0h[i];
            b1[i] = w1l[i]; b1[8 + i] = w1h[i];
        }
        acc0 = wmma_bf16(a, b0, acc0);
        acc1 = wmma_bf16(a, b1, acc1);
    }

#pragma unroll
    for (int r = 0; r < 8; ++r) {
        const int row = tm * 16 + r + g * 8;
        const int b2  = row / LCONV;
        const int l2  = row % LCONV;
        Out[(size_t)b2 * ostride + (size_t)lr * LCONV + l2] =
            (__bf16)(acc0[r] + bias[lr]);
        Out[(size_t)b2 * ostride + (size_t)(16 + lr) * LCONV + l2] =
            (__bf16)(acc1[r] + bias[16 + lr]);
    }
}

// ---------------------------------------------------------------------------
// Graph helpers (monotone float<->uint encoding for atomic segment-max)
// ---------------------------------------------------------------------------
__device__ __forceinline__ unsigned fenc(float f) {
    unsigned u = __float_as_uint(f);
    return (u & 0x80000000u) ? ~u : (u | 0x80000000u);
}
__device__ __forceinline__ float fdec(unsigned e) {
    unsigned u = (e & 0x80000000u) ? (e & 0x7FFFFFFFu) : ~e;
    return __uint_as_float(u);
}

__global__ void k_att(const float* __restrict__ hbuf,
                      const float* __restrict__ att_s,
                      const float* __restrict__ att_d,
                      float* __restrict__ as_, float* __restrict__ ad_, int N)
{
    int t = blockIdx.x * blockDim.x + threadIdx.x;
    if (t >= N * HEADS) return;
    int n = t / HEADS, h = t % HEADS;
    const float* hp = hbuf + (size_t)n * FH + h * NODES_F0;
    float s0 = 0.f, s1 = 0.f;
    for (int f = 0; f < NODES_F0; ++f) {
        s0 += hp[f] * att_s[h * NODES_F0 + f];
        s1 += hp[f] * att_d[h * NODES_F0 + f];
    }
    as_[t] = s0; ad_[t] = s1;
}

__global__ void k_edge_max(const int* __restrict__ ei,
                           const float* __restrict__ as_, const float* __restrict__ ad_,
                           unsigned* __restrict__ menc, float* __restrict__ deg,
                           int E, int N)
{
    int idx = blockIdx.x * blockDim.x + threadIdx.x;
    if (idx >= E + N) return;
    int s = (idx < E) ? ei[idx]     : (idx - E);
    int d = (idx < E) ? ei[E + idx] : (idx - E);
    atomicAdd(&deg[d], 1.0f);
#pragma unroll
    for (int h = 0; h < HEADS; ++h) {
        float e = as_[s * HEADS + h] + ad_[d * HEADS + h];
        e = (e > 0.f) ? e : 0.2f * e;
        atomicMax(&menc[(size_t)d * HEADS + h], fenc(e));
    }
}

__global__ void k_edge_den(const int* __restrict__ ei,
                           const float* __restrict__ as_, const float* __restrict__ ad_,
                           const unsigned* __restrict__ menc,
                           float* __restrict__ den, float* __restrict__ exbuf,
                           int E, int N)
{
    int idx = blockIdx.x * blockDim.x + threadIdx.x;
    if (idx >= E + N) return;
    int s = (idx < E) ? ei[idx]     : (idx - E);
    int d = (idx < E) ? ei[E + idx] : (idx - E);
#pragma unroll
    for (int h = 0; h < HEADS; ++h) {
        float e = as_[s * HEADS + h] + ad_[d * HEADS + h];
        e = (e > 0.f) ? e : 0.2f * e;
        float ex = __expf(e - fdec(menc[(size_t)d * HEADS + h]));
        exbuf[(size_t)idx * HEADS + h] = ex;
        atomicAdd(&den[(size_t)d * HEADS + h], ex);
    }
}

__global__ void k_edge_agg(const int* __restrict__ ei,
                           const float* __restrict__ hbuf,
                           const float* __restrict__ den,
                           const float* __restrict__ exbuf,
                           float* __restrict__ x1, int E, int N)
{
    int idx = blockIdx.x;
    int s = (idx < E) ? ei[idx]     : (idx - E);
    int d = (idx < E) ? ei[E + idx] : (idx - E);
    for (int j = threadIdx.x; j < FH; j += blockDim.x) {
        int h = j / NODES_F0;
        float coef = exbuf[(size_t)idx * HEADS + h] / den[(size_t)d * HEADS + h];
        atomicAdd(&x1[(size_t)d * FH + j], coef * hbuf[(size_t)s * FH + j]);
    }
}

__global__ void k_gcn_agg(const int* __restrict__ ei,
                          const float* __restrict__ xw,
                          const float* __restrict__ deg,
                          float* __restrict__ x2, int E, int N)
{
    int idx = blockIdx.x;
    int s = (idx < E) ? ei[idx]     : (idx - E);
    int d = (idx < E) ? ei[E + idx] : (idx - E);
    float norm = rsqrtf(deg[s]) * rsqrtf(deg[d]);
    for (int j = threadIdx.x; j < FH; j += blockDim.x) {
        atomicAdd(&x2[(size_t)d * FH + j], norm * xw[(size_t)s * FH + j]);
    }
}

__global__ void k_bias_bn_relu(float* __restrict__ x,
                               const float* __restrict__ bias,
                               const float* __restrict__ bnp, int M, int C)
{
    size_t i = (size_t)blockIdx.x * blockDim.x + threadIdx.x;
    if (i >= (size_t)M * C) return;
    int c = (int)(i % C);
    float v = x[i] + bias[c];
    v = bnp[c] * (v - bnp[2 * C + c]) * rsqrtf(bnp[3 * C + c] + 1e-5f) + bnp[C + c];
    x[i] = fmaxf(v, 0.f);
}

__global__ void k_pool(const float* __restrict__ x1, const float* __restrict__ x2,
                       const int* __restrict__ batch,
                       float* __restrict__ psum, float* __restrict__ pcnt, int N)
{
    int n = blockIdx.x;
    int b = batch[n];
    if (threadIdx.x == 0) atomicAdd(&pcnt[b], 1.0f);
    for (int j = threadIdx.x; j < 2 * FH; j += blockDim.x) {
        float v = (j < FH) ? x1[(size_t)n * FH + j] : x2[(size_t)n * FH + (j - FH)];
        atomicAdd(&psum[(size_t)b * (2 * FH) + j], v);
    }
}

__global__ void k_pool_div(const float* __restrict__ psum,
                           const float* __restrict__ pcnt, float* __restrict__ g)
{
    int i = blockIdx.x * blockDim.x + threadIdx.x;
    if (i >= BATCHES * 2 * FH) return;
    int b = i / (2 * FH);
    g[i] = psum[i] / fmaxf(pcnt[b], 1.0f);
}

__global__ void k_concat(const float* __restrict__ g2, const float* __restrict__ xt,
                         float* __restrict__ xc)
{
    int i = blockIdx.x * blockDim.x + threadIdx.x;
    if (i >= BATCHES * 256) return;
    int b = i >> 8, c = i & 255;
    xc[i] = (c < 128) ? g2[b * 128 + c] : xt[b * 128 + (c - 128)];
}

__global__ void k_out(const float* __restrict__ f2, const float* __restrict__ w,
                      const float* __restrict__ b, float* __restrict__ out)
{
    int bidx = blockIdx.x * blockDim.x + threadIdx.x;
    if (bidx >= BATCHES) return;
    float s = 0.f;
    for (int k = 0; k < 512; ++k) s += f2[bidx * 512 + k] * w[k];
    out[bidx] = s + b[0];
}

// ---------------------------------------------------------------------------
extern "C" void kernel_launch(void* const* d_in, const int* in_sizes, int n_in,
                              void* d_out, int out_size, void* d_ws, size_t ws_size,
                              hipStream_t stream)
{
    const float* x       = (const float*)d_in[0];
    const int*   ei      = (const int*)  d_in[1];
    const int*   batch   = (const int*)  d_in[2];
    const float* target  = (const float*)d_in[3];
    const float* w_gat   = (const float*)d_in[4];
    const float* att_src = (const float*)d_in[5];
    const float* att_dst = (const float*)d_in[6];
    const float* b_gat   = (const float*)d_in[7];
    const float* w_gcn   = (const float*)d_in[8];
    const float* b_gcn   = (const float*)d_in[9];
    const float* bn1     = (const float*)d_in[10];
    const float* bn2     = (const float*)d_in[11];
    const float* w_fcg1  = (const float*)d_in[12];
    const float* b_fcg1  = (const float*)d_in[13];
    const float* bn3     = (const float*)d_in[14];
    const float* w_fcg2  = (const float*)d_in[15];
    const float* b_fcg2  = (const float*)d_in[16];
    const float* bn4     = (const float*)d_in[17];
    const float* w_conv  = (const float*)d_in[18];
    const float* b_conv  = (const float*)d_in[19];
    const float* w_fcxt  = (const float*)d_in[20];
    const float* b_fcxt  = (const float*)d_in[21];
    const float* bn6     = (const float*)d_in[22];
    const float* w_fc1   = (const float*)d_in[23];
    const float* b_fc1   = (const float*)d_in[24];
    const float* bn7     = (const float*)d_in[25];
    const float* w_fc2   = (const float*)d_in[26];
    const float* b_fc2   = (const float*)d_in[27];
    const float* bn8     = (const float*)d_in[28];
    const float* w_out   = (const float*)d_in[29];
    const float* b_out   = (const float*)d_in[30];

    const int N  = in_sizes[0] / NODES_F0;   // 16384
    const int E  = in_sizes[1] / 2;          // 98304
    const int EN = E + N;

    auto cdiv = [](int a, int b) { return (a + b - 1) / b; };
    auto pad32 = [&](int k) { return cdiv(k, 32) * 32; };
    auto pad64 = [&](int k) { return cdiv(k, 64) * 64; };

    const int Kp78   = pad64(NODES_F0);   // 128
    const int Kp780  = pad64(FH);         // 832
    const int Kp1560 = pad64(2 * FH);     // 1600
    const int Kp1500 = pad64(1500);       // 1536
    const int KpConv = pad64(CONVFLAT);   // 32576
    const int Np780  = pad32(FH);         // 800
    const int Np1500 = pad32(1500);       // 1504

    // ---- workspace layout ----
    char* wp = (char*)d_ws;
    auto alloc = [&](size_t nbytes) -> char* {
        char* p = wp; wp += (nbytes + 255) & ~(size_t)255; return p;
    };
    float*    hbuf  = (float*)   alloc(sizeof(float) * (size_t)N * FH);
    float*    x1    = (float*)   alloc(sizeof(float) * (size_t)N * FH);
    float*    xw    = (float*)   alloc(sizeof(float) * (size_t)N * FH);
    float*    x2    = (float*)   alloc(sizeof(float) * (size_t)N * FH);
    float*    as_   = (float*)   alloc(sizeof(float) * (size_t)N * HEADS);
    float*    ad_   = (float*)   alloc(sizeof(float) * (size_t)N * HEADS);
    unsigned* menc  = (unsigned*)alloc(sizeof(unsigned) * (size_t)N * HEADS);
    float*    den   = (float*)   alloc(sizeof(float) * (size_t)N * HEADS);
    float*    deg   = (float*)   alloc(sizeof(float) * (size_t)N);
    float*    exbuf = (float*)   alloc(sizeof(float) * (size_t)EN * HEADS);
    float*    psum  = (float*)   alloc(sizeof(float) * BATCHES * 2 * FH);
    float*    pcnt  = (float*)   alloc(sizeof(float) * BATCHES);
    float*    gbuf  = (float*)   alloc(sizeof(float) * BATCHES * 2 * FH);
    float*    g1    = (float*)   alloc(sizeof(float) * BATCHES * 1500);
    float*    g2    = (float*)   alloc(sizeof(float) * BATCHES * 128);
    float*    xt    = (float*)   alloc(sizeof(float) * BATCHES * 128);
    float*    xc    = (float*)   alloc(sizeof(float) * BATCHES * 256);
    float*    f1    = (float*)   alloc(sizeof(float) * BATCHES * 1024);
    float*    f2    = (float*)   alloc(sizeof(float) * BATCHES * 512);
    // bf16 staging
    __bf16* xP      = (__bf16*)alloc(2 * (size_t)N * Kp78);
    __bf16* wgatT   = (__bf16*)alloc(2 * (size_t)Np780 * Kp78);
    __bf16* x1P     = (__bf16*)alloc(2 * (size_t)N * Kp780);
    __bf16* wgcnT   = (__bf16*)alloc(2 * (size_t)Np780 * Kp780);
    __bf16* gP      = (__bf16*)alloc(2 * (size_t)BATCHES * Kp1560);
    __bf16* wfcg1T  = (__bf16*)alloc(2 * (size_t)Np1500 * Kp1560);
    __bf16* g1P     = (__bf16*)alloc(2 * (size_t)BATCHES * Kp1500);
    __bf16* wfcg2T  = (__bf16*)alloc(2 * (size_t)128 * Kp1500);
    __bf16* wconvT  = (__bf16*)alloc(2 * (size_t)32 * KFLAT);
    __bf16* convA   = (__bf16*)alloc(2 * (size_t)BATCHES * KpConv);
    __bf16* wfcxtT  = (__bf16*)alloc(2 * (size_t)128 * KpConv);
    __bf16* xcP     = (__bf16*)alloc(2 * (size_t)BATCHES * 256);
    __bf16* wfc1T   = (__bf16*)alloc(2 * (size_t)1024 * 256);
    __bf16* f1P     = (__bf16*)alloc(2 * (size_t)BATCHES * 1024);
    __bf16* wfc2T   = (__bf16*)alloc(2 * (size_t)512 * 1024);

    // ---- zero accumulators / padded regions ----
    hipMemsetAsync(x1,    0, sizeof(float) * (size_t)N * FH, stream);
    hipMemsetAsync(x2,    0, sizeof(float) * (size_t)N * FH, stream);
    hipMemsetAsync(menc,  0, sizeof(unsigned) * (size_t)N * HEADS, stream);
    hipMemsetAsync(den,   0, sizeof(float) * (size_t)N * HEADS, stream);
    hipMemsetAsync(deg,   0, sizeof(float) * (size_t)N, stream);
    hipMemsetAsync(psum,  0, sizeof(float) * BATCHES * 2 * FH, stream);
    hipMemsetAsync(pcnt,  0, sizeof(float) * BATCHES, stream);
    hipMemsetAsync(convA, 0, 2 * (size_t)BATCHES * KpConv, stream);

    auto pack_a = [&](const float* A, __bf16* Ap, int M, int K, int Kp) {
        size_t tot = (size_t)M * Kp;
        k_pack_a<<<(int)((tot + 255) / 256), 256, 0, stream>>>(A, Ap, M, K, Kp);
    };
    auto pack_bt = [&](const float* B, __bf16* Btp, int K, int Nn, int Np, int Kp) {
        size_t tot = (size_t)Np * Kp;
        k_pack_bt<<<(int)((tot + 255) / 256), 256, 0, stream>>>(B, Btp, K, Nn, Np, Kp);
    };
    auto gemm = [&](const __bf16* Ap, const __bf16* Btp, float* C,
                    int M, int Nn, int Kp, const float* bias, const float* bn, int epi) {
        int Mtiles = cdiv(M, 16);
        dim3 grid(cdiv(cdiv(Nn, 16), 2), cdiv(Mtiles, 4));
        k_wmma_gemm<<<grid, dim3(128), 0, stream>>>(Ap, Btp, C, M, Nn, Kp, Mtiles,
                                                    bias, bn, epi);
    };

    // ---- pack all weights ----
    pack_bt(w_gat,  wgatT,  NODES_F0, FH,   Np780,  Kp78);
    pack_bt(w_gcn,  wgcnT,  FH,       FH,   Np780,  Kp780);
    pack_bt(w_fcg1, wfcg1T, 2 * FH,   1500, Np1500, Kp1560);
    pack_bt(w_fcg2, wfcg2T, 1500,     128,  128,    Kp1500);
    pack_a (w_conv, wconvT, 32, KFLAT, KFLAT);        // already [32, 5120] = B^T
    pack_bt(w_fcxt, wfcxtT, CONVFLAT, 128,  128,    KpConv);
    pack_bt(w_fc1,  wfc1T,  256,      1024, 1024,   256);
    pack_bt(w_fc2,  wfc2T,  1024,     512,  512,    1024);

    // ---- GAT: h = x @ w_gat ----
    pack_a(x, xP, N, NODES_F0, Kp78);
    gemm(xP, wgatT, hbuf, N, FH, Kp78, nullptr, nullptr, 0);
    k_att<<<(N * HEADS + 255) / 256, 256, 0, stream>>>(hbuf, att_src, att_dst, as_, ad_, N);
    k_edge_max<<<(EN + 255) / 256, 256, 0, stream>>>(ei, as_, ad_, menc, deg, E, N);
    k_edge_den<<<(EN + 255) / 256, 256, 0, stream>>>(ei, as_, ad_, menc, den, exbuf, E, N);
    k_edge_agg<<<EN, 256, 0, stream>>>(ei, hbuf, den, exbuf, x1, E, N);
    k_bias_bn_relu<<<(int)(((size_t)N * FH + 255) / 256), 256, 0, stream>>>(x1, b_gat, bn1, N, FH);

    // ---- GCN ----
    pack_a(x1, x1P, N, FH, Kp780);
    gemm(x1P, wgcnT, xw, N, FH, Kp780, nullptr, nullptr, 0);
    k_gcn_agg<<<EN, 256, 0, stream>>>(ei, xw, deg, x2, E, N);
    k_bias_bn_relu<<<(int)(((size_t)N * FH + 255) / 256), 256, 0, stream>>>(x2, b_gcn, bn2, N, FH);

    // ---- global mean pool + graph MLP ----
    k_pool<<<N, 256, 0, stream>>>(x1, x2, batch, psum, pcnt, N);
    k_pool_div<<<(BATCHES * 2 * FH + 255) / 256, 256, 0, stream>>>(psum, pcnt, gbuf);
    pack_a(gbuf, gP, BATCHES, 2 * FH, Kp1560);
    gemm(gP, wfcg1T, g1, BATCHES, 1500, Kp1560, b_fcg1, bn3, 1);
    pack_a(g1, g1P, BATCHES, 1500, Kp1500);
    gemm(g1P, wfcg2T, g2, BATCHES, 128, Kp1500, b_fcg2, bn4, 1);

    // ---- protein branch: Conv1d (implicit GEMM, writes padded bf16) + fcxt ----
    k_conv_wmma<<<dim3((BATCHES * LCONV / 16) / 4), dim3(128), 0, stream>>>(
        target, wconvT, b_conv, convA, KpConv);
    gemm(convA, wfcxtT, xt, BATCHES, 128, KpConv, b_fcxt, bn6, 2);

    // ---- fusion head ----
    k_concat<<<(BATCHES * 256 + 255) / 256, 256, 0, stream>>>(g2, xt, xc);
    pack_a(xc, xcP, BATCHES, 256, 256);
    gemm(xcP, wfc1T, f1, BATCHES, 1024, 256, b_fc1, bn7, 2);
    pack_a(f1, f1P, BATCHES, 1024, 1024);
    gemm(f1P, wfc2T, f2, BATCHES, 512, 1024, b_fc2, bn8, 2);
    k_out<<<(BATCHES + 127) / 128, 128, 0, stream>>>(f2, w_out, b_out, (float*)d_out);
}
